// Graph_Encoder_9320079033176
// MI455X (gfx1250) — matrix operand order repaired and checked by reading, not compiled, not run on hardware
//
#include <hip/hip_runtime.h>
#include <hip/hip_bf16.h>

typedef __attribute__((ext_vector_type(16))) _Float16 v16h;
typedef __attribute__((ext_vector_type(8)))  _Float16 v8h;
typedef __attribute__((ext_vector_type(8)))  float    v8f;

#define N_NODES 50000
#define N_EDGES 800000
#define FDIM    256     // HEADS*HID and F_IN
#define HEADS   4
#define HID     64
#define NEG_SLOPE 0.2f

// LDS tile: 256 rows (N) x 256 f16 (K), padded to 264 f16/row so that the
// 16-lane b128 fragment reads start at banks 4*l mod 64 (conflict-free).
#define LDS_ROW 264
#define TILES_PER_BLOCK 8   // m-tiles consumed per block (4 iterations x 2)

// ---------- helpers ----------
__device__ __forceinline__ unsigned fkey(float v) {
    unsigned u = __float_as_uint(v);
    return (u & 0x80000000u) ? ~u : (u | 0x80000000u);
}
__device__ __forceinline__ float fdecode(unsigned k) {
    return __uint_as_float((k & 0x80000000u) ? (k ^ 0x80000000u) : ~k);
}

// ---------- utility kernels ----------
__global__ void zero_u32_kernel(unsigned* __restrict__ p, int n) {
    int i = blockIdx.x * blockDim.x + threadIdx.x;
    if (i < n) p[i] = 0u;
}

__global__ void cvt_f32_to_f16_kernel(const float* __restrict__ in,
                                      _Float16* __restrict__ out, int n) {
    int i = blockIdx.x * blockDim.x + threadIdx.x;
    if (i < n) out[i] = (_Float16)in[i];
}

// W[k][n] (256x256 row-major) -> Wt[n][k] f16 (column-major of W)
__global__ void transpose_w_f16_kernel(const float* __restrict__ W,
                                       _Float16* __restrict__ Wt) {
    int i = blockIdx.x * blockDim.x + threadIdx.x;   // 65536 threads
    int k = i >> 8;
    int n = i & 255;
    Wt[n * FDIM + k] = (_Float16)W[i];
}

// ---------- WMMA GEMM: C[M,256] = A[M,256](f16,row-major) @ B(Wt[n][k] f16) ----------
// Whole Wt (128KB) is staged to LDS once per block; each block then runs
// TILES_PER_BLOCK m-tiles (8 waves: 2 m-tiles x 4 n-groups per iteration),
// with A fragments from global (b128) and B fragments from ds_load_b128.
__global__ void __launch_bounds__(256)
gemm_wmma_f16_kernel(const _Float16* __restrict__ A,
                     const _Float16* __restrict__ Bt,
                     float* __restrict__ C, int M) {
    __shared__ _Float16 lw[256 * LDS_ROW];

    const int tid = threadIdx.x;

    // cooperative fill: 65536 f16 = 8192 uint4, 32 per thread (padded rows)
    for (int j = tid; j < 8192; j += 256) {
        int row = j >> 5;
        int c   = j & 31;
        ((uint4*)(lw + row * LDS_ROW))[c] = ((const uint4*)(Bt + row * FDIM))[c];
    }
    __syncthreads();

    const int wave = tid >> 5;        // 0..7
    const int lane = tid & 31;
    const int lid  = lane & 15;
    const int hlf  = lane >> 4;       // 0 or 1
    const int ngroup = wave & 3;      // 64 columns per wave
    const int nbase  = ngroup * 64;

    const int mt0 = blockIdx.x * TILES_PER_BLOCK + (wave >> 2);

    for (int it = 0; it < TILES_PER_BLOCK / 2; ++it) {
        const int mtile = mt0 + 2 * it;
        const int mbase = mtile * 16;
        if (mbase >= M) break;        // wave-uniform: EXEC stays all-1 for WMMA

        v8f acc0 = {}, acc1 = {}, acc2 = {}, acc3 = {};

        const _Float16* arow = A + (size_t)(mbase + lid) * FDIM;
        const _Float16* lb   = lw + (size_t)(nbase + lid) * LDS_ROW + hlf * 16;

        #pragma unroll
        for (int kb = 0; kb < FDIM; kb += 32) {
            union { v16h v; v8h h[2]; } a, b0, b1, b2, b3;
            a.h[0] = *(const v8h*)(arow + kb + hlf * 16);
            a.h[1] = *(const v8h*)(arow + kb + hlf * 16 + 8);

            b0.h[0] = *(const v8h*)(lb + kb);
            b0.h[1] = *(const v8h*)(lb + kb + 8);
            b1.h[0] = *(const v8h*)(lb + 16 * LDS_ROW + kb);
            b1.h[1] = *(const v8h*)(lb + 16 * LDS_ROW + kb + 8);
            b2.h[0] = *(const v8h*)(lb + 32 * LDS_ROW + kb);
            b2.h[1] = *(const v8h*)(lb + 32 * LDS_ROW + kb + 8);
            b3.h[0] = *(const v8h*)(lb + 48 * LDS_ROW + kb);
            b3.h[1] = *(const v8h*)(lb + 48 * LDS_ROW + kb + 8);

            acc0 = __builtin_amdgcn_wmma_f32_16x16x32_f16(false, a.v, false, b0.v, (short)0, acc0, false, false);
            acc1 = __builtin_amdgcn_wmma_f32_16x16x32_f16(false, a.v, false, b1.v, (short)0, acc1, false, false);
            acc2 = __builtin_amdgcn_wmma_f32_16x16x32_f16(false, a.v, false, b2.v, (short)0, acc2, false, false);
            acc3 = __builtin_amdgcn_wmma_f32_16x16x32_f16(false, a.v, false, b3.v, (short)0, acc3, false, false);
        }

        // store: VGPR r -> row mbase + 8*hlf + r, col nbase + t*16 + lid
        float* crow = C + (size_t)(mbase + hlf * 8) * FDIM + nbase + lid;
        #pragma unroll
        for (int r = 0; r < 8; ++r) {
            crow[(size_t)r * FDIM + 0]  = acc0[r];
            crow[(size_t)r * FDIM + 16] = acc1[r];
            crow[(size_t)r * FDIM + 32] = acc2[r];
            crow[(size_t)r * FDIM + 48] = acc3[r];
        }
    }
}

// ---------- per-node attention coefficients el/er ----------
__global__ void node_attn_kernel(const float* __restrict__ feat,
                                 const float* __restrict__ a_l,
                                 const float* __restrict__ a_r,
                                 float* __restrict__ el,
                                 float* __restrict__ er, int N) {
    int i = blockIdx.x * blockDim.x + threadIdx.x;   // (n,h)
    if (i >= N * HEADS) return;
    int n = i >> 2, h = i & 3;
    const float4* f  = (const float4*)(feat + (size_t)n * FDIM + h * HID);
    const float4* al = (const float4*)(a_l + h * HID);
    const float4* ar = (const float4*)(a_r + h * HID);
    float sl = 0.f, sr = 0.f;
    #pragma unroll
    for (int j = 0; j < HID / 4; ++j) {
        float4 fv = f[j], lv = al[j], rv = ar[j];
        sl += fv.x * lv.x + fv.y * lv.y + fv.z * lv.z + fv.w * lv.w;
        sr += fv.x * rv.x + fv.y * rv.y + fv.z * rv.z + fv.w * rv.w;
    }
    el[i] = sl;
    er[i] = sr;
}

// ---------- edge phase 1: logits + segment max ----------
__global__ void edge_logits_kernel(const int* __restrict__ src,
                                   const int* __restrict__ dst,
                                   const float* __restrict__ el,
                                   const float* __restrict__ er,
                                   float* __restrict__ ebuf,
                                   unsigned* __restrict__ mkey, int E) {
    int e = blockIdx.x * blockDim.x + threadIdx.x;
    if (e >= E) return;
    int s = src[e], d = dst[e];
    #pragma unroll
    for (int h = 0; h < HEADS; ++h) {
        float v = el[s * HEADS + h] + er[d * HEADS + h];
        v = (v > 0.f) ? v : NEG_SLOPE * v;            // leaky_relu
        ebuf[e * HEADS + h] = v;
        atomicMax(&mkey[d * HEADS + h], fkey(v));
    }
}

// ---------- edge phase 2: exp(e - m) + segment sum ----------
__global__ void edge_exp_kernel(const int* __restrict__ dst,
                                const unsigned* __restrict__ mkey,
                                float* __restrict__ ebuf,
                                float* __restrict__ ssum, int E) {
    int i = blockIdx.x * blockDim.x + threadIdx.x;   // e*4+h
    if (i >= E * HEADS) return;
    int e = i >> 2, h = i & 3;
    int d = dst[e];
    float m  = fdecode(mkey[d * HEADS + h]);
    float ex = __expf(ebuf[i] - m);
    ebuf[i] = ex;
    unsafeAtomicAdd(&ssum[d * HEADS + h], ex);
}

// ---------- edge phase 3: weighted scatter-add aggregation ----------
// One 256-thread block covers the full 256-dim message; loops over a chunk of edges.
__global__ void __launch_bounds__(256)
edge_aggregate_kernel(const int* __restrict__ src,
                      const int* __restrict__ dst,
                      const float* __restrict__ feat,
                      const float* __restrict__ ebuf,
                      const float* __restrict__ ssum,
                      float* __restrict__ rst,
                      int E, int edges_per_block) {
    int d = threadIdx.x;         // 0..255
    int h = d >> 6;              // head for this dim
    int e0 = blockIdx.x * edges_per_block;
    int e1 = e0 + edges_per_block; if (e1 > E) e1 = E;
    for (int e = e0; e < e1; ++e) {
        int s = src[e], t = dst[e];
        float denom = ssum[t * HEADS + h];
        float alpha = ebuf[e * HEADS + h] / fmaxf(denom, 1e-9f);
        unsafeAtomicAdd(&rst[(size_t)t * FDIM + d],
                        feat[(size_t)s * FDIM + d] * alpha);
    }
}

// ---------- epilogues ----------
// layer 0: bias + ELU -> f16 activations for next GEMM
__global__ void finalize_f16_kernel(const float* __restrict__ rst,
                                    const float* __restrict__ b,
                                    _Float16* __restrict__ out_h, int N) {
    int i = blockIdx.x * blockDim.x + threadIdx.x;   // n*256 + d
    if (i >= N * FDIM) return;
    int d = i & 255;
    float v = rst[i] + b[d];
    v = (v > 0.f) ? v : (__expf(v) - 1.0f);          // elu, alpha=1
    out_h[i] = (_Float16)v;
}

// layer 1: bias + ELU -> head-major final output: out[h*N*HID + n*HID + dd]
__global__ void finalize_out_kernel(const float* __restrict__ rst,
                                    const float* __restrict__ b,
                                    float* __restrict__ out, int N) {
    int i = blockIdx.x * blockDim.x + threadIdx.x;   // n*256 + d
    if (i >= N * FDIM) return;
    int n = i >> 8;
    int d = i & 255;
    int h = d >> 6;
    int dd = d & 63;
    float v = rst[i] + b[d];
    v = (v > 0.f) ? v : (__expf(v) - 1.0f);
    out[((size_t)h * N + n) * HID + dd] = v;
}

// ---------- host launcher ----------
extern "C" void kernel_launch(void* const* d_in, const int* in_sizes, int n_in,
                              void* d_out, int out_size, void* d_ws, size_t ws_size,
                              hipStream_t stream) {
    const float* x   = (const float*)d_in[0];
    const int*   src = (const int*)d_in[1];
    const int*   dst = (const int*)d_in[2];
    const float* W0  = (const float*)d_in[3];
    const float* al0 = (const float*)d_in[4];
    const float* ar0 = (const float*)d_in[5];
    const float* b0  = (const float*)d_in[6];
    const float* W1  = (const float*)d_in[7];
    const float* al1 = (const float*)d_in[8];
    const float* ar1 = (const float*)d_in[9];
    const float* b1  = (const float*)d_in[10];

    const int N = N_NODES, E = N_EDGES;

    // workspace carve-out (256B aligned regions)
    char* ws = (char*)d_ws;
    size_t off = 0;
    auto take = [&](size_t bytes) -> char* {
        char* p = ws + off;
        off = (off + bytes + 255) & ~(size_t)255;
        return p;
    };
    _Float16* xf16  = (_Float16*)take((size_t)N * FDIM * 2);  // activations (f16), reused for layer 1
    _Float16* wtf16 = (_Float16*)take((size_t)FDIM * FDIM * 2);
    float*    feat  = (float*)take((size_t)N * FDIM * 4);
    float*    rst   = (float*)take((size_t)N * FDIM * 4);
    float*    el    = (float*)take((size_t)N * HEADS * 4);
    float*    er    = (float*)take((size_t)N * HEADS * 4);
    unsigned* mkey  = (unsigned*)take((size_t)N * HEADS * 4);
    float*    ssum  = (float*)take((size_t)N * HEADS * 4);
    float*    ebuf  = (float*)take((size_t)E * HEADS * 4);

    const int T = 256;
    const int nf_grid   = (N * FDIM + T - 1) / T;       // 50000
    const int nh_grid   = (N * HEADS + T - 1) / T;
    const int e_grid    = (E + T - 1) / T;
    const int eh_grid   = (E * HEADS + T - 1) / T;
    const int mtiles    = N / 16;                       // 3125 (exact)
    const int gemm_grid = (mtiles + TILES_PER_BLOCK - 1) / TILES_PER_BLOCK;  // 391
    const int epb       = 8;                            // edges per aggregation block
    const int agg_grid  = (E + epb - 1) / epb;

    // input activations -> f16
    cvt_f32_to_f16_kernel<<<nf_grid, T, 0, stream>>>(x, xf16, N * FDIM);

    // ================= layer 0 =================
    transpose_w_f16_kernel<<<(FDIM * FDIM) / T, T, 0, stream>>>(W0, wtf16);
    gemm_wmma_f16_kernel<<<gemm_grid, T, 0, stream>>>(xf16, wtf16, feat, N);
    node_attn_kernel<<<nh_grid, T, 0, stream>>>(feat, al0, ar0, el, er, N);
    zero_u32_kernel<<<nf_grid, T, 0, stream>>>((unsigned*)rst, N * FDIM);
    zero_u32_kernel<<<nh_grid, T, 0, stream>>>(mkey, N * HEADS);
    zero_u32_kernel<<<nh_grid, T, 0, stream>>>((unsigned*)ssum, N * HEADS);
    edge_logits_kernel<<<e_grid, T, 0, stream>>>(src, dst, el, er, ebuf, mkey, E);
    edge_exp_kernel<<<eh_grid, T, 0, stream>>>(dst, mkey, ebuf, ssum, E);
    edge_aggregate_kernel<<<agg_grid, T, 0, stream>>>(src, dst, feat, ebuf, ssum, rst, E, epb);
    finalize_f16_kernel<<<nf_grid, T, 0, stream>>>(rst, b0, xf16, N);

    // ================= layer 1 =================
    transpose_w_f16_kernel<<<(FDIM * FDIM) / T, T, 0, stream>>>(W1, wtf16);
    gemm_wmma_f16_kernel<<<gemm_grid, T, 0, stream>>>(xf16, wtf16, feat, N);
    node_attn_kernel<<<nh_grid, T, 0, stream>>>(feat, al1, ar1, el, er, N);
    zero_u32_kernel<<<nf_grid, T, 0, stream>>>((unsigned*)rst, N * FDIM);
    zero_u32_kernel<<<nh_grid, T, 0, stream>>>(mkey, N * HEADS);
    zero_u32_kernel<<<nh_grid, T, 0, stream>>>((unsigned*)ssum, N * HEADS);
    edge_logits_kernel<<<e_grid, T, 0, stream>>>(src, dst, el, er, ebuf, mkey, E);
    edge_exp_kernel<<<eh_grid, T, 0, stream>>>(dst, mkey, ebuf, ssum, E);
    edge_aggregate_kernel<<<agg_grid, T, 0, stream>>>(src, dst, feat, ebuf, ssum, rst, E, epb);
    finalize_out_kernel<<<nf_grid, T, 0, stream>>>(rst, b1, (float*)d_out, N);
}
